// MultiHeadAttention_33191507263681
// MI455X (gfx1250) — matrix operand order repaired and checked by reading, not compiled, not run on hardware
//
#include <hip/hip_runtime.h>
#include <hip/hip_bf16.h>

typedef __attribute__((ext_vector_type(16))) _Float16 v16h;
typedef __attribute__((ext_vector_type(8)))  _Float16 v8h;
typedef __attribute__((ext_vector_type(8)))  float    v8f;
typedef __attribute__((ext_vector_type(4)))  float    v4f;

#define S_LEN  4096
#define DMODEL 1024
#define NHEAD  16
#define HDIM   64

// ---------------------------------------------------------------------------
// WMMA wrapper: D(16x16 f32) = A(16x32 f16) * B(32x16 f16) + C
// ---------------------------------------------------------------------------
__device__ __forceinline__ v8f wmma16x16x32(v16h a, v16h b, v8f c) {
  return __builtin_amdgcn_wmma_f32_16x16x32_f16(
      /*neg_a=*/false, a, /*neg_b=*/false, b,
      /*c_mod=*/(short)0, c, /*reuse_a=*/false, /*reuse_b=*/false);
}

// ---------------------------------------------------------------------------
// A-operand fragment (16x32, 16-bit): lane holds row m = lane&15.
// half h = lane>>4: v16h elems 0..15 = K[h*16 .. h*16+15]
// `row` must already point at  base + m*ld + k0.
// ---------------------------------------------------------------------------
__device__ __forceinline__ v16h fragA_f16(const _Float16* row, int h) {
  const _Float16* p = row + h * 16;
  v8h lo = *(const v8h*)(p);
  v8h hi = *(const v8h*)(p + 8);
  v16h r;
#pragma unroll
  for (int i = 0; i < 8; ++i) { r[i] = lo[i]; r[i + 8] = hi[i]; }
  return r;
}

__device__ __forceinline__ v16h fragA_f32(const float* row, int h) {
  const float* p = row + h * 16;
  v4f a0 = *(const v4f*)(p);
  v4f a1 = *(const v4f*)(p + 4);
  v4f b0 = *(const v4f*)(p + 8);
  v4f b1 = *(const v4f*)(p + 12);
  v16h r;
#pragma unroll
  for (int i = 0; i < 4; ++i) {
    r[i]      = (_Float16)a0[i];
    r[i + 4]  = (_Float16)a1[i];
    r[i + 8]  = (_Float16)b0[i];
    r[i + 12] = (_Float16)b1[i];
  }
  return r;
}

// ---------------------------------------------------------------------------
// B-operand fragment (32x16, 16-bit): lane holds column n = lane&15, i.e. a
// row of B^T with K contiguous: elems 0..15 = K[h*16 .. h*16+15].
// `rowT` points at  Bt_base + n*ld + k0   (row of B-transpose).
// ---------------------------------------------------------------------------
__device__ __forceinline__ v16h fragB_f16(const _Float16* rowT, int h) {
  return *(const v16h*)(rowT + h * 16);
}

// intra-half (16 lane) reductions: halves stay disjoint for xor masks < 16
__device__ __forceinline__ float half16_max(float x) {
#pragma unroll
  for (int off = 8; off >= 1; off >>= 1) x = fmaxf(x, __shfl_xor(x, off, 32));
  return x;
}
__device__ __forceinline__ float half16_sum(float x) {
#pragma unroll
  for (int off = 8; off >= 1; off >>= 1) x += __shfl_xor(x, off, 32);
  return x;
}

// ---------------------------------------------------------------------------
// Kernel 0: f32 -> f16 conversion (weights, done once per launch)
// ---------------------------------------------------------------------------
__global__ void mha_cvt_f16_kernel(const float* __restrict__ src,
                                   _Float16* __restrict__ dst, int n) {
  int i = blockIdx.x * blockDim.x + threadIdx.x;
  if (i < n) dst[i] = (_Float16)src[i];
}

// ---------------------------------------------------------------------------
// Kernel 1: projection GEMM  C = X @ W^T + b
//   X : [S, D] f32,  Wh : [D, D] f16 (row-major, row = out-feature)
//   One wave computes a 32x64 tile of C (2 row-halves reuse the B fragments).
//   TRANSPOSE==0 : dst[h][s][dk]   (Q / K layout)
//   TRANSPOSE==1 : dst[h][dv][s]   (V^T layout)
//   Compile-time layout switch -> straight-line (branch-free) store epilogue.
// ---------------------------------------------------------------------------
template <int TRANSPOSE>
__global__ __launch_bounds__(32) void mha_proj_kernel(
    const float* __restrict__ X, const _Float16* __restrict__ Wh,
    const float* __restrict__ bias, _Float16* __restrict__ dst) {
  const int lane = threadIdx.x;
  const int n  = lane & 15;
  const int hh = lane >> 4;
  const int col0 = blockIdx.x * 64;   // 16 column-groups
  const int row0 = blockIdx.y * 32;   // 128 row-blocks

  v8f acc[2][4];
#pragma unroll
  for (int t = 0; t < 4; ++t) {
    float bv = bias[col0 + 16 * t + n];
#pragma unroll
    for (int r = 0; r < 8; ++r) { acc[0][t][r] = bv; acc[1][t][r] = bv; }
  }

  const float* arow0 = X + (size_t)(row0 + n) * DMODEL;
  const float* arow1 = X + (size_t)(row0 + 16 + n) * DMODEL;
  for (int k0 = 0; k0 < DMODEL; k0 += 32) {
    v16h af0 = fragA_f32(arow0 + k0, hh);
    v16h af1 = fragA_f32(arow1 + k0, hh);
#pragma unroll
    for (int t = 0; t < 4; ++t) {
      v16h bf = fragB_f16(Wh + (size_t)(col0 + 16 * t + n) * DMODEL + k0, hh);
      acc[0][t] = wmma16x16x32(af0, bf, acc[0][t]);
      acc[1][t] = wmma16x16x32(af1, bf, acc[1][t]);
    }
  }

  // C layout: VGPR r, half hh -> row m = r + 8*hh ; column = lane&15 (+16t)
#pragma unroll
  for (int u = 0; u < 2; ++u)
#pragma unroll
    for (int t = 0; t < 4; ++t)
#pragma unroll
      for (int r = 0; r < 8; ++r) {
        int s    = row0 + 16 * u + r + 8 * hh;
        int col  = col0 + 16 * t + n;
        int head = col >> 6;
        int dd   = col & 63;
        _Float16 val = (_Float16)acc[u][t][r];
        if (TRANSPOSE)
          dst[((size_t)head * HDIM + dd) * S_LEN + s] = val;   // [H][dv][S]
        else
          dst[((size_t)head * S_LEN + s) * HDIM + dd] = val;   // [H][S][dk]
      }
}

// ---------------------------------------------------------------------------
// Kernel 2: causal flash attention.
//   Workgroup = 4 waves covering 64 queries of one head; each 32-key K/V tile
//   is staged into LDS once per workgroup and consumed by all 4 waves
//   (4x less L2 K/V traffic than per-wave streaming).
//   Qh,Kh : [H][S][64] f16,  Vt : [H][64][S] f16,  O : [S][H*64] f16
// ---------------------------------------------------------------------------
__global__ __launch_bounds__(128) void mha_attn_kernel(
    const _Float16* __restrict__ Qh, const _Float16* __restrict__ Kh,
    const _Float16* __restrict__ Vt, _Float16* __restrict__ O) {
  __shared__ __align__(32) _Float16 ldsK[32 * 64];       // 32 keys x 64 dk
  __shared__ __align__(32) _Float16 ldsV[64 * 32];       // 64 dv   x 32 keys
  __shared__ __align__(32) _Float16 ldsP[4][16 * 32];    // per-wave P restage

  const int tid  = threadIdx.x;
  const int lane = tid & 31;
  const int wave = tid >> 5;
  const int n  = lane & 15;
  const int hh = lane >> 4;
  const int qwg  = blockIdx.x * 64;       // 64-query block for the workgroup
  const int q0   = qwg + wave * 16;       // this wave's 16 queries
  const int head = blockIdx.y;

  const _Float16* Qb = Qh + (size_t)head * S_LEN * HDIM;
  const _Float16* Kb = Kh + (size_t)head * S_LEN * HDIM;
  const _Float16* Vb = Vt + (size_t)head * HDIM * S_LEN;

  // resident Q fragments (row = q0 + n), two K-chunks of 32
  v16h qf0 = fragA_f16(Qb + (size_t)(q0 + n) * HDIM + 0, hh);
  v16h qf1 = fragA_f16(Qb + (size_t)(q0 + n) * HDIM + 32, hh);

  v8f oacc[4];
#pragma unroll
  for (int t = 0; t < 4; ++t)
#pragma unroll
    for (int r = 0; r < 8; ++r) oacc[t][r] = 0.0f;

  float mrun[8], lrun[8];
#pragma unroll
  for (int r = 0; r < 8; ++r) { mrun[r] = -3.0e38f; lrun[r] = 0.0f; }

  const float scale = 0.125f;    // 1/sqrt(64)
  const int kend = qwg + 64;     // uniform across the workgroup (causal)

  // staging addresses for this thread (32B chunk each for K and V tiles)
  const int krow = tid >> 2, kcol = (tid & 3) * 16;   // 32 rows x 64 halves
  const int vrow = tid >> 1, vcol = (tid & 1) * 16;   // 64 rows x 32 halves

  for (int kb = 0; kb < kend; kb += 32) {
    // ---- cooperative stage of K and V^T tiles into LDS ----
    *(v16h*)(ldsK + tid * 16) =
        *(const v16h*)(Kb + (size_t)(kb + krow) * HDIM + kcol);
    *(v16h*)(ldsV + tid * 16) =
        *(const v16h*)(Vb + (size_t)vrow * S_LEN + kb + vcol);
    if (kb + 32 < kend) {  // prefetch next tile -> global_prefetch_b8
      __builtin_prefetch(Kb + (size_t)(kb + 32 + krow) * HDIM + kcol, 0, 1);
      __builtin_prefetch(Vb + (size_t)vrow * S_LEN + kb + 32 + vcol, 0, 1);
    }
    __syncthreads();

    if (kb < q0 + 16) {  // wave-uniform: skip tiles fully above the diagonal
      // ---- S = Q K^T : two 16x16 score tiles over 32 keys ----
      v8f s0 = {}, s1 = {};
      {
        v16h b0 = fragB_f16(ldsK + (size_t)n * HDIM + 0, hh);
        v16h b1 = fragB_f16(ldsK + (size_t)n * HDIM + 32, hh);
        s0 = wmma16x16x32(qf0, b0, s0);
        s0 = wmma16x16x32(qf1, b1, s0);
        v16h c0 = fragB_f16(ldsK + (size_t)(16 + n) * HDIM + 0, hh);
        v16h c1 = fragB_f16(ldsK + (size_t)(16 + n) * HDIM + 32, hh);
        s1 = wmma16x16x32(qf0, c0, s1);
        s1 = wmma16x16x32(qf1, c1, s1);
      }

      // ---- scale + causal mask ----
      const bool diag = (kb + 32 > q0);
#pragma unroll
      for (int r = 0; r < 8; ++r) {
        int m = q0 + r + 8 * hh;
        float v0 = s0[r] * scale;
        float v1 = s1[r] * scale;
        if (diag) {
          if (kb + n      > m) v0 = -3.0e38f;
          if (kb + 16 + n > m) v1 = -3.0e38f;
        }
        s0[r] = v0; s1[r] = v1;
      }

      // ---- online softmax ----
      float newm[8], alpha[8], rs[8];
#pragma unroll
      for (int r = 0; r < 8; ++r) {
        float bm = half16_max(fmaxf(s0[r], s1[r]));
        newm[r]  = fmaxf(mrun[r], bm);
        alpha[r] = __expf(mrun[r] - newm[r]);
        float p0 = __expf(s0[r] - newm[r]);
        float p1 = __expf(s1[r] - newm[r]);
        s0[r] = p0; s1[r] = p1;
        rs[r] = half16_sum(p0 + p1);
      }
#pragma unroll
      for (int r = 0; r < 8; ++r) {
        lrun[r] = lrun[r] * alpha[r] + rs[r];
        mrun[r] = newm[r];
      }
#pragma unroll
      for (int t = 0; t < 4; ++t)
#pragma unroll
        for (int r = 0; r < 8; ++r) oacc[t][r] *= alpha[r];

      // ---- restage P (C layout -> A layout) through private LDS slice ----
      _Float16* myP = &ldsP[wave][0];
#pragma unroll
      for (int r = 0; r < 8; ++r) {
        int m = r + 8 * hh;
        myP[m * 32 + n]      = (_Float16)s0[r];
        myP[m * 32 + 16 + n] = (_Float16)s1[r];
      }
      asm volatile("s_wait_dscnt 0" ::: "memory");  // intra-wave LDS RAW

      v16h pf = fragA_f16(myP + n * 32, hh);        // P as A operand (16x32)

      // ---- O += P V : B operand = rows of V^T (LDS tile) ----
#pragma unroll
      for (int t = 0; t < 4; ++t) {
        v16h vf = fragB_f16(ldsV + (size_t)(16 * t + n) * 32, hh);
        oacc[t] = wmma16x16x32(pf, vf, oacc[t]);
      }
    }
    __syncthreads();  // all waves done with ldsK/ldsV before next stage
  }

  // ---- epilogue: normalize and store f16 [S][H*64] ----
  float inv[8];
#pragma unroll
  for (int r = 0; r < 8; ++r) inv[r] = 1.0f / lrun[r];
#pragma unroll
  for (int t = 0; t < 4; ++t)
#pragma unroll
    for (int r = 0; r < 8; ++r) {
      int s = q0 + r + 8 * hh;
      O[(size_t)s * DMODEL + head * HDIM + 16 * t + n] =
          (_Float16)(oacc[t][r] * inv[r]);
    }
}

// ---------------------------------------------------------------------------
// Kernel 3: output projection  out = O @ Wo^T + bo   (all-f16 operands)
//   One wave computes a 32x64 tile (B fragments reused across 2 row-halves).
// ---------------------------------------------------------------------------
__global__ __launch_bounds__(32) void mha_oproj_kernel(
    const _Float16* __restrict__ A, const _Float16* __restrict__ WoH,
    const float* __restrict__ bo, float* __restrict__ out) {
  const int lane = threadIdx.x;
  const int n  = lane & 15;
  const int hh = lane >> 4;
  const int col0 = blockIdx.x * 64;
  const int row0 = blockIdx.y * 32;

  v8f acc[2][4];
#pragma unroll
  for (int t = 0; t < 4; ++t) {
    float bv = bo[col0 + 16 * t + n];
#pragma unroll
    for (int r = 0; r < 8; ++r) { acc[0][t][r] = bv; acc[1][t][r] = bv; }
  }

  const _Float16* arow0 = A + (size_t)(row0 + n) * DMODEL;
  const _Float16* arow1 = A + (size_t)(row0 + 16 + n) * DMODEL;
  for (int k0 = 0; k0 < DMODEL; k0 += 32) {
    v16h af0 = fragA_f16(arow0 + k0, hh);
    v16h af1 = fragA_f16(arow1 + k0, hh);
#pragma unroll
    for (int t = 0; t < 4; ++t) {
      v16h bf = fragB_f16(WoH + (size_t)(col0 + 16 * t + n) * DMODEL + k0, hh);
      acc[0][t] = wmma16x16x32(af0, bf, acc[0][t]);
      acc[1][t] = wmma16x16x32(af1, bf, acc[1][t]);
    }
  }

#pragma unroll
  for (int u = 0; u < 2; ++u)
#pragma unroll
    for (int t = 0; t < 4; ++t)
#pragma unroll
      for (int r = 0; r < 8; ++r)
        out[(size_t)(row0 + 16 * u + r + 8 * hh) * DMODEL + col0 + 16 * t + n] =
            acc[u][t][r];
}

// ---------------------------------------------------------------------------
extern "C" void kernel_launch(void* const* d_in, const int* in_sizes, int n_in,
                              void* d_out, int out_size, void* d_ws,
                              size_t ws_size, hipStream_t stream) {
  const float* q  = (const float*)d_in[0];
  const float* k  = (const float*)d_in[1];
  const float* v  = (const float*)d_in[2];
  const float* Wq = (const float*)d_in[3];
  const float* bq = (const float*)d_in[4];
  const float* Wk = (const float*)d_in[5];
  const float* bk = (const float*)d_in[6];
  const float* Wv = (const float*)d_in[7];
  const float* bv = (const float*)d_in[8];
  const float* Wo = (const float*)d_in[9];
  const float* bo = (const float*)d_in[10];
  float* out = (float*)d_out;

  const size_t MB = 1024ull * 1024ull;
  char* ws = (char*)d_ws;
  _Float16* WqH = (_Float16*)(ws + 0 * MB);    // 2 MB each
  _Float16* WkH = (_Float16*)(ws + 2 * MB);
  _Float16* WvH = (_Float16*)(ws + 4 * MB);
  _Float16* WoH = (_Float16*)(ws + 6 * MB);
  _Float16* Qh  = (_Float16*)(ws + 8 * MB);    // [H][S][64]  8 MB
  _Float16* Kh  = (_Float16*)(ws + 16 * MB);   // [H][S][64]  8 MB
  _Float16* Vt  = (_Float16*)(ws + 24 * MB);   // [H][64][S]  8 MB
  _Float16* Oa  = (_Float16*)(ws + 32 * MB);   // [S][1024]   8 MB

  const int WN = DMODEL * DMODEL;  // 1M elems per weight matrix
  dim3 cvtGrid((WN + 255) / 256), cvtBlk(256);
  mha_cvt_f16_kernel<<<cvtGrid, cvtBlk, 0, stream>>>(Wq, WqH, WN);
  mha_cvt_f16_kernel<<<cvtGrid, cvtBlk, 0, stream>>>(Wk, WkH, WN);
  mha_cvt_f16_kernel<<<cvtGrid, cvtBlk, 0, stream>>>(Wv, WvH, WN);
  mha_cvt_f16_kernel<<<cvtGrid, cvtBlk, 0, stream>>>(Wo, WoH, WN);

  dim3 wave(32);
  dim3 gemmGrid(DMODEL / 64, S_LEN / 32);  // (16, 128)
  mha_proj_kernel<0><<<gemmGrid, wave, 0, stream>>>(q, WqH, bq, Qh);
  mha_proj_kernel<0><<<gemmGrid, wave, 0, stream>>>(k, WkH, bk, Kh);
  mha_proj_kernel<1><<<gemmGrid, wave, 0, stream>>>(v, WvH, bv, Vt);

  dim3 attnGrid(S_LEN / 64, NHEAD);  // (64, 16)
  dim3 attnBlk(128);                 // 4 waves share K/V LDS tiles
  mha_attn_kernel<<<attnGrid, attnBlk, 0, stream>>>(Qh, Kh, Vt, Oa);

  mha_oproj_kernel<<<gemmGrid, wave, 0, stream>>>(Oa, WoH, bo, out);
}
